// classifier_selective_ts_with_g_90099823936040
// MI455X (gfx1250) — compile-verified, matrix operand order checked by reading, not run on hardware
//
#include <hip/hip_runtime.h>

// ---------------------------------------------------------------------------
// classifier_selective_ts_with_g on MI455X (gfx1250, wave32, WMMA)
// GEMMs: f16 x f16 -> f32 accumulate via v_wmma_f32_16x16x32_f16.
// Tile movement: GLOBAL_LOAD_ASYNC_TO_LDS_B128 (ASYNCcnt), double-buffered LDS.
// ---------------------------------------------------------------------------

#define MV      19
#define DIMV    768
#define DV      384
#define ROWLEN  (MV * DIMV)          // 14592
#define LDS_PITCH 40                 // 32 halves + 8 pad (bank stagger, 80B rows)
#define TILE_HALVES (128 * LDS_PITCH)        // 5120 halves = 10240 B per tile
#define BUF_HALVES  (2 * TILE_HALVES)        // A + B tiles per buffer
#define BUF_BYTES   (BUF_HALVES * 2)         // 20480 B

typedef __attribute__((ext_vector_type(16))) _Float16 v16h;
typedef __attribute__((ext_vector_type(8)))  float    v8f;

union FragU { v16h v; uint2 d[4]; };

// Load a 16x32 (A) / 32x16 (B) f16 fragment from LDS following the CDNA5
// 16-bit layout: lanes 0-15 hold K=0..7 then K=16..23; lanes 16-31 hold
// K=8..15 then K=24..31 (kq0 = 0 or 2 in uint2 units).
__device__ __forceinline__ void load_frag(const _Float16* base, int kq0, FragU& f) {
    const uint2* p = (const uint2*)base;
    f.d[0] = p[kq0 + 0];
    f.d[1] = p[kq0 + 1];
    f.d[2] = p[kq0 + 4];
    f.d[3] = p[kq0 + 5];
}

// ---------------------------------------------------------------------------
// Generic tiled WMMA GEMM:  C = epi(A1 @ W1^T [+ A2 @ W2^T] + bias)
//   A : rows x K f16, row stride lda, batch stride aStr
//   W : N x K    f16, row stride ldw, batch stride wStr
//   C : rows x N, row stride ldc, batch stride cStr (f16 or f32)
// EPI: 0 = +bias ; 2 = relu(+bias) * (scale/sqrt(1+eps)) + shift  (eval BN)
// Block: 256 threads = 8 waves; tile 128(M) x 128(N) x 32(K);
// wave (wm,wn) in 4x2 grid computes 32x64 = 2x4 WMMA accumulators.
// ---------------------------------------------------------------------------
template<int EPI, bool OUTF32, bool HAS2>
__global__ __launch_bounds__(256)
void k_gemm(const _Float16* __restrict__ A1, int lda1, long long aStr1,
            const _Float16* __restrict__ W1, int ldw1, long long wStr1, int K1,
            const _Float16* __restrict__ A2, int lda2, long long aStr2,
            const _Float16* __restrict__ W2, int ldw2, long long wStr2, int K2,
            const float* __restrict__ bias, long long bStr,
            const float* __restrict__ scale, const float* __restrict__ shift,
            void* __restrict__ Cv, int ldc, long long cStr, int Nn)
{
    const int bz = blockIdx.z;
    A1 += (size_t)bz * aStr1;
    W1 += (size_t)bz * wStr1;
    if (HAS2) { A2 += (size_t)bz * aStr2; W2 += (size_t)bz * wStr2; }
    bias += (size_t)bz * bStr;

    const int mBase = blockIdx.y * 128;
    const int nBase = blockIdx.x * 128;

    // double-buffered: [buf][A tile | B tile]
    __shared__ _Float16 smem[2 * BUF_HALVES];
    const unsigned ldsBase = (unsigned)(uintptr_t)(&smem[0]);   // low 32b = LDS offset

    const int t      = threadIdx.x;
    const int lane   = t & 31;
    const int waveId = t >> 5;
    const int wm     = waveId >> 1;   // 0..3 -> 32-row strip
    const int wn     = waveId & 1;    // 0..1 -> 64-col strip
    const int lrow   = lane & 15;
    const int kq0    = (lane >> 4) * 2;   // uint2 index of first K chunk

    // tile loader mapping: 256 threads cover 64 rows x 4 k-chunks, twice
    const int tr = t >> 2;            // 0..63
    const int tk = (t & 3) << 3;      // 0,8,16,24 (halves)

    const int steps1 = K1 >> 5;
    const int totalSteps = steps1 + (HAS2 ? (K2 >> 5) : 0);

    // issue one 128x32 A-tile + 128x32 W-tile as 4 async b128 loads per thread
    auto issue = [&](int s) {
        const _Float16* Ab; const _Float16* Wb; int lda, ldw, k0;
        if (!HAS2 || s < steps1) { Ab = A1; lda = lda1; Wb = W1; ldw = ldw1; k0 = s << 5; }
        else                     { Ab = A2; lda = lda2; Wb = W2; ldw = ldw2; k0 = (s - steps1) << 5; }
        const unsigned bufByte = (unsigned)((s & 1) * BUF_BYTES);
#pragma unroll
        for (int h = 0; h < 2; ++h) {
            const int row = tr + h * 64;
            const unsigned la = ldsBase + bufByte + (unsigned)((row * LDS_PITCH + tk) * 2);
            const _Float16* ga = Ab + (size_t)(mBase + row) * lda + k0 + tk;
            asm volatile("global_load_async_to_lds_b128 %0, %1, off"
                         :: "v"(la), "v"((unsigned long long)(uintptr_t)ga)
                         : "memory");
            // clamp OOB W-rows (cols >= Nn are store-masked; keeps ASYNCcnt uniform)
            int n = nBase + row; if (n >= Nn) n = Nn - 1;
            const unsigned lb = la + (unsigned)(TILE_HALVES * 2);
            const _Float16* gw = Wb + (size_t)n * ldw + k0 + tk;
            asm volatile("global_load_async_to_lds_b128 %0, %1, off"
                         :: "v"(lb), "v"((unsigned long long)(uintptr_t)gw)
                         : "memory");
        }
    };

    v8f acc[2][4];
#pragma unroll
    for (int i = 0; i < 2; ++i)
#pragma unroll
        for (int j = 0; j < 4; ++j)
            acc[i][j] = (v8f){0.f,0.f,0.f,0.f,0.f,0.f,0.f,0.f};

    issue(0);
    for (int s = 0; s < totalSteps; ++s) {
        if (s + 1 < totalSteps) {
            issue(s + 1);                                  // prefetch next tile
            asm volatile("s_wait_asynccnt 0x4" ::: "memory");  // tile s done (in-order)
        } else {
            asm volatile("s_wait_asynccnt 0x0" ::: "memory");
        }
        __syncthreads();

        const _Float16* sAb = smem + (s & 1) * BUF_HALVES;
        const _Float16* sBb = sAb + TILE_HALVES;

        FragU fa[2], fb[4];
#pragma unroll
        for (int i = 0; i < 2; ++i)
            load_frag(&sAb[(wm * 32 + i * 16 + lrow) * LDS_PITCH], kq0, fa[i]);
#pragma unroll
        for (int j = 0; j < 4; ++j)
            load_frag(&sBb[(wn * 64 + j * 16 + lrow) * LDS_PITCH], kq0, fb[j]);

#pragma unroll
        for (int i = 0; i < 2; ++i)
#pragma unroll
            for (int j = 0; j < 4; ++j)
                acc[i][j] = __builtin_amdgcn_wmma_f32_16x16x32_f16(
                    false, fa[i].v, false, fb[j].v,
                    (short)0, acc[i][j], false, false);

        __syncthreads();   // all reads of this buffer done before it is re-filled
    }

    // epilogue + store; C/D layout: lane n=lrow, vgpr r -> row r (+8 for hi lanes)
    const float kBN = 0.9999950000374997f;   // 1/sqrt(1 + 1e-5)
    float*     Cf = (float*)Cv;
    _Float16*  Ch = (_Float16*)Cv;
    const size_t coff = (size_t)bz * (size_t)cStr;
    const int rhi = (lane >> 4) << 3;        // +8 for lanes 16..31

#pragma unroll
    for (int i = 0; i < 2; ++i)
#pragma unroll
        for (int j = 0; j < 4; ++j) {
            const int col = nBase + wn * 64 + j * 16 + lrow;
            if (col < Nn) {
                const float b = bias[col];
                float sc = 0.f, sh = 0.f;
                if (EPI == 2) { sc = scale[col] * kBN; sh = shift[col]; }
#pragma unroll
                for (int r = 0; r < 8; ++r) {
                    const int grow = mBase + wm * 32 + i * 16 + r + rhi;
                    float v = acc[i][j][r] + b;
                    if (EPI == 2) v = fmaxf(v, 0.f) * sc + sh;
                    const size_t idx = coff + (size_t)grow * ldc + col;
                    if (OUTF32) Cf[idx] = v;
                    else        Ch[idx] = (_Float16)v;
                }
            }
        }
}

// ---------------------------------------------------------------------------
// fp32 -> f16 conversion (weights + g)
// ---------------------------------------------------------------------------
__global__ void k_cvt(const float* __restrict__ in, _Float16* __restrict__ out, int n) {
    int i = blockIdx.x * 256 + threadIdx.x;
    if (i < n) out[i] = (_Float16)in[i];
}

// fea_u[b,c] = sum_m x[b, m*768 + c]  (f16 out, feeds GEMM1)
__global__ void k_colsum(const float* __restrict__ x, _Float16* __restrict__ out) {
    const int idx = blockIdx.x * 256 + threadIdx.x;
    const int b = idx / DIMV;
    const int c = idx - b * DIMV;
    const float* p = x + (size_t)b * ROWLEN + c;
    float s = 0.f;
#pragma unroll
    for (int m = 0; m < MV; ++m) s += p[m * DIMV];
    out[idx] = (_Float16)s;
}

// softmax over the 19 branches + weighted reduction of x -> fea_v (f16)
__global__ void k_softmax_feav(const _Float16* __restrict__ att,
                               const float* __restrict__ x,
                               _Float16* __restrict__ feav) {
    const int idx = blockIdx.x * 256 + threadIdx.x;
    const int b = idx / DIMV;
    const int o = idx - b * DIMV;
    const _Float16* ap = att + (size_t)b * ROWLEN + o;
    float a[MV];
    float mx = -3.0e38f;
#pragma unroll
    for (int m = 0; m < MV; ++m) { a[m] = (float)ap[m * DIMV]; mx = fmaxf(mx, a[m]); }
    float sum = 0.f;
#pragma unroll
    for (int m = 0; m < MV; ++m) { a[m] = __expf(a[m] - mx); sum += a[m]; }
    const float inv = 1.f / sum;
    const float* xp = x + (size_t)b * ROWLEN + o;
    float v = 0.f;
#pragma unroll
    for (int m = 0; m < MV; ++m) v += xp[m * DIMV] * a[m];
    feav[idx] = (_Float16)(v * inv);
}

// ---------------------------------------------------------------------------
extern "C" void kernel_launch(void* const* d_in, const int* in_sizes, int n_in,
                              void* d_out, int out_size, void* d_ws, size_t ws_size,
                              hipStream_t stream) {
    const float* x      = (const float*)d_in[0];
    const float* g      = (const float*)d_in[1];
    const float* fc1_W  = (const float*)d_in[2];
    const float* fc1_b  = (const float*)d_in[3];
    const float* bn1_g  = (const float*)d_in[4];
    const float* bn1_b  = (const float*)d_in[5];
    const float* fc2_W  = (const float*)d_in[6];
    const float* fc2_b  = (const float*)d_in[7];
    const float* fcs_W  = (const float*)d_in[8];
    const float* fcs_b  = (const float*)d_in[9];
    const float* gfcs_W = (const float*)d_in[10];
    const float* gfcs_b = (const float*)d_in[11];
    const float* l1_W   = (const float*)d_in[12];
    const float* l1_b   = (const float*)d_in[13];
    const float* bna_g  = (const float*)d_in[14];
    const float* bna_b  = (const float*)d_in[15];
    const float* l2_W   = (const float*)d_in[16];
    const float* l2_b   = (const float*)d_in[17];
    const float* bnb_g  = (const float*)d_in[18];
    const float* bnb_b  = (const float*)d_in[19];
    const float* l3_W   = (const float*)d_in[20];
    const float* l3_b   = (const float*)d_in[21];
    float* out = (float*)d_out;

    const int rows = in_sizes[0] / ROWLEN;          // 4096
    const int mB   = rows / 128;                    // 32 row-blocks

    // ---- workspace carve (f16 element counts) ----
    char* ws = (char*)d_ws;
    size_t off = 0;
    auto alloc = [&](size_t nElems) -> _Float16* {
        _Float16* p = (_Float16*)(ws + off);
        off += (nElems * sizeof(_Float16) + 255) & ~(size_t)255;
        return p;
    };
    _Float16* g16    = alloc((size_t)rows * ROWLEN);
    _Float16* fc1h   = alloc((size_t)DV * DIMV);
    _Float16* fc2h   = alloc((size_t)DV * DV);
    _Float16* fcsh   = alloc((size_t)MV * DV * DV);
    _Float16* gfcsh  = alloc((size_t)MV * DIMV * (DIMV + DV));
    _Float16* l1h    = alloc((size_t)128 * DIMV);
    _Float16* l2h    = alloc((size_t)32 * 128);
    _Float16* l3h    = alloc((size_t)20 * 32);
    _Float16* feau   = alloc((size_t)rows * DIMV);
    _Float16* h1     = alloc((size_t)rows * DV);
    _Float16* feaz   = alloc((size_t)rows * DV);
    _Float16* vec16  = alloc((size_t)rows * MV * DV);
    _Float16* att16  = alloc((size_t)rows * ROWLEN);
    _Float16* feav   = alloc((size_t)rows * DIMV);
    _Float16* ha     = alloc((size_t)rows * 128);
    _Float16* hb     = alloc((size_t)rows * 32);

    auto cvt = [&](const float* src, _Float16* dst, size_t n) {
        k_cvt<<<(unsigned)((n + 255) / 256), 256, 0, stream>>>(src, dst, (int)n);
    };
    cvt(g,      g16,   (size_t)rows * ROWLEN);
    cvt(fc1_W,  fc1h,  (size_t)DV * DIMV);
    cvt(fc2_W,  fc2h,  (size_t)DV * DV);
    cvt(fcs_W,  fcsh,  (size_t)MV * DV * DV);
    cvt(gfcs_W, gfcsh, (size_t)MV * DIMV * (DIMV + DV));
    cvt(l1_W,   l1h,   (size_t)128 * DIMV);
    cvt(l2_W,   l2h,   (size_t)32 * 128);
    cvt(l3_W,   l3h,   (size_t)20 * 32);

    // 1) fea_u = sum_m x
    k_colsum<<<(unsigned)((size_t)rows * DIMV / 256), 256, 0, stream>>>(x, feau);

    // 2) h1 = BN(relu(fea_u @ fc1^T + b))
    k_gemm<2, false, false><<<dim3(3, mB, 1), 256, 0, stream>>>(
        feau, DIMV, 0, fc1h, DIMV, 0, DIMV,
        nullptr, 0, 0, nullptr, 0, 0, 0,
        fc1_b, 0, bn1_g, bn1_b,
        (void*)h1, DV, 0, DV);

    // 3) fea_z = h1 @ fc2^T + b
    k_gemm<0, false, false><<<dim3(3, mB, 1), 256, 0, stream>>>(
        h1, DV, 0, fc2h, DV, 0, DV,
        nullptr, 0, 0, nullptr, 0, 0, 0,
        fc2_b, 0, nullptr, nullptr,
        (void*)feaz, DV, 0, DV);

    // 4) vec[:,m,:] = fea_z @ fcs[m]^T + b   (batched over 19)
    k_gemm<0, false, false><<<dim3(3, mB, MV), 256, 0, stream>>>(
        feaz, DV, 0, fcsh, DV, (long long)DV * DV, DV,
        nullptr, 0, 0, nullptr, 0, 0, 0,
        fcs_b, DV, nullptr, nullptr,
        (void*)vec16, MV * DV, DV, DV);

    // 5) att[:,m,:] = vec_m @ gfcs[m][:, :384]^T + g_m @ gfcs[m][:, 384:]^T + b
    //    (concat handled as a second K-accumulation; dominant GEMM, 137.7 GFLOP)
    k_gemm<0, false, true><<<dim3(6, mB, MV), 256, 0, stream>>>(
        vec16, MV * DV, DV, gfcsh, DIMV + DV, (long long)DIMV * (DIMV + DV), DV,
        g16, ROWLEN, DIMV, gfcsh + DV, DIMV + DV, (long long)DIMV * (DIMV + DV), DIMV,
        gfcs_b, DIMV, nullptr, nullptr,
        (void*)att16, ROWLEN, DIMV, DIMV);

    // 6) softmax over m + fea_v = sum_m x * att
    k_softmax_feav<<<(unsigned)((size_t)rows * DIMV / 256), 256, 0, stream>>>(att16, x, feav);

    // 7) head: 768 -> 128 (relu+BN) -> 32 (relu+BN) -> 20 (f32 out)
    k_gemm<2, false, false><<<dim3(1, mB, 1), 256, 0, stream>>>(
        feav, DIMV, 0, l1h, DIMV, 0, DIMV,
        nullptr, 0, 0, nullptr, 0, 0, 0,
        l1_b, 0, bna_g, bna_b,
        (void*)ha, 128, 0, 128);

    k_gemm<2, false, false><<<dim3(1, mB, 1), 256, 0, stream>>>(
        ha, 128, 0, l2h, 128, 0, 128,
        nullptr, 0, 0, nullptr, 0, 0, 0,
        l2_b, 0, bnb_g, bnb_b,
        (void*)hb, 32, 0, 32);

    k_gemm<0, true, false><<<dim3(1, mB, 1), 256, 0, stream>>>(
        hb, 32, 0, l3h, 32, 0, 32,
        nullptr, 0, 0, nullptr, 0, 0, 0,
        l3_b, 0, nullptr, nullptr,
        (void*)out, 20, 0, 20);
}
